// StyledConvBlock_34617436406293
// MI455X (gfx1250) — matrix operand order repaired and checked by reading, not compiled, not run on hardware
//
#include <hip/hip_runtime.h>

typedef __bf16 bf16_t;
typedef __attribute__((ext_vector_type(16))) bf16_t v16bf;
typedef __attribute__((ext_vector_type(8)))  float  v8f;
typedef __attribute__((ext_vector_type(4)))  unsigned int u32x4;

#define N_    32
#define C_    128
#define H_    64
#define W_    64
#define HW_   4096
#define TASKD 512
#define HP_   66         // padded height (h+1, halo rows 0 and 65)
#define WP_   66         // padded width  (w+1, halo cols 0 and 65)

// padded activation elements: 32*66*66*128 = 17,842,176 bf16 = 35,684,352 bytes
#define XTP_ELEMS (N_ * HP_ * WP_ * C_)

__device__ __forceinline__ unsigned short f2bf(float f) {
    unsigned int u = __float_as_uint(f);
    u += 0x7FFFu + ((u >> 16) & 1u);   // round-to-nearest-even
    return (unsigned short)(u >> 16);
}

// ---------------------------------------------------------------------------
// Zero the padded activation buffer (establishes the zero halo every launch)
// ---------------------------------------------------------------------------
__global__ void k_zero(u32x4* __restrict__ p, int n16) {
    int i = blockIdx.x * 256 + threadIdx.x;
    if (i < n16) p[i] = (u32x4)0u;
}

// ---------------------------------------------------------------------------
// x (N,C,H,W) f32 -> xTp (N,66,66,C) bf16 interior  (LDS transpose)
// ---------------------------------------------------------------------------
__global__ void k_convert_x(const float* __restrict__ x, unsigned short* __restrict__ xTp) {
    __shared__ float tile[C_ * W_];            // 32 KB
    int b = blockIdx.x;                        // n*64 + h
    int n = b >> 6, h = b & 63;
    const float* src = x + (size_t)n * C_ * HW_ + h * W_;
    for (int i = threadIdx.x; i < C_ * W_; i += 256) {
        int c = i >> 6, w = i & 63;
        tile[(c << 6) | w] = src[(size_t)c * HW_ + w];
    }
    __syncthreads();
    unsigned short* dst = xTp + (((size_t)(n * HP_ + h + 1) * WP_ + 1) << 7);
    for (int i = threadIdx.x; i < C_ * W_; i += 256) {
        int w = i >> 7, c = i & 127;
        dst[((size_t)w << 7) + c] = f2bf(tile[(c << 6) + w]);
    }
}

// ---------------------------------------------------------------------------
// Small linears: bi1/bi2 = b + task@tbw.T*st + tbb ; s1/s2 = style@adw.T*st + adb
// ---------------------------------------------------------------------------
__global__ void k_small(const float* __restrict__ task, const float* __restrict__ style,
                        const float* __restrict__ b1, const float* __restrict__ tb1_w, const float* __restrict__ tb1_b,
                        const float* __restrict__ b2, const float* __restrict__ tb2_w, const float* __restrict__ tb2_b,
                        const float* __restrict__ ad1_w, const float* __restrict__ ad1_b,
                        const float* __restrict__ ad2_w, const float* __restrict__ ad2_b,
                        float* __restrict__ bi1, float* __restrict__ bi2,
                        float* __restrict__ s1, float* __restrict__ s2) {
    const float st = 0.0625f;                  // sqrt(2/512)
    int g = blockIdx.x * 256 + threadIdx.x;
    if (g < 4096) {
        int n = g >> 7, c = g & 127;
        const float* tr = task + n * TASKD; const float* wr = tb1_w + (size_t)c * TASKD;
        float acc = 0.f;
        for (int k = 0; k < TASKD; k++) acc += tr[k] * wr[k];
        bi1[g] = b1[c] + acc * st + tb1_b[c];
    } else if (g < 8192) {
        int g2 = g - 4096; int n = g2 >> 7, c = g2 & 127;
        const float* tr = task + n * TASKD; const float* wr = tb2_w + (size_t)c * TASKD;
        float acc = 0.f;
        for (int k = 0; k < TASKD; k++) acc += tr[k] * wr[k];
        bi2[g2] = b2[c] + acc * st + tb2_b[c];
    } else if (g < 16384) {
        int g2 = g - 8192; int n = g2 >> 8, j = g2 & 255;
        const float* tr = style + n * TASKD; const float* wr = ad1_w + (size_t)j * TASKD;
        float acc = 0.f;
        for (int k = 0; k < TASKD; k++) acc += tr[k] * wr[k];
        s1[g2] = acc * st + ad1_b[j];
    } else {
        int g2 = g - 16384; int n = g2 >> 8, j = g2 & 255;
        const float* tr = style + n * TASKD; const float* wr = ad2_w + (size_t)j * TASKD;
        float acc = 0.f;
        for (int k = 0; k < TASKD; k++) acc += tr[k] * wr[k];
        s2[g2] = acc * st + ad2_b[j];
    }
}

// ---------------------------------------------------------------------------
// Fused AdaFM: gamma/beta = task @ t_w.T * st + t_b, then
// Wm[n][tap][co][ci] = bf16((W[co][ci][tap]*gamma + beta) * sqrt(2/1152))
// ---------------------------------------------------------------------------
__global__ void __launch_bounds__(256) k_modw(const float* __restrict__ task,
                                              const float* __restrict__ t_w,
                                              const float* __restrict__ t_b,
                                              const float* __restrict__ Wc,
                                              unsigned short* __restrict__ Wm) {
    __shared__ float tk[32 * 64];              // 8 KB task chunk
    int tid = threadIdx.x;
    int ci = tid & 127;
    int co = (blockIdx.x << 1) + (tid >> 7);
    int rowg = ((co << 7) + ci) << 1;          // gamma row; beta = rowg+1
    const float* gw = t_w + (size_t)rowg * TASKD;
    const float* bw = gw + TASKD;
    float accg[32], accb[32];
#pragma unroll
    for (int n = 0; n < 32; n++) { accg[n] = 0.f; accb[n] = 0.f; }
    for (int kc = 0; kc < TASKD; kc += 64) {
        for (int i = tid; i < 2048; i += 256) {
            int n = i >> 6, k = i & 63;
            tk[i] = task[n * TASKD + kc + k];
        }
        __syncthreads();
        for (int k = 0; k < 64; k++) {
            float gv = gw[kc + k], bv = bw[kc + k];
#pragma unroll
            for (int n = 0; n < 32; n++) {
                float t = tk[(n << 6) + k];
                accg[n] += t * gv;
                accb[n] += t * bv;
            }
        }
        __syncthreads();
    }
    const float st = 0.0625f;                  // sqrt(2/512)
    const float wsc = 0.041666666666666664f;   // sqrt(2/1152) = 1/24
    float w9[9];
#pragma unroll
    for (int t9 = 0; t9 < 9; t9++) w9[t9] = Wc[(size_t)((co << 7) + ci) * 9 + t9];
    float tbg = t_b[rowg], tbb = t_b[rowg + 1];
    for (int n = 0; n < 32; n++) {
        float gamma = accg[n] * st + tbg;
        float beta  = accb[n] * st + tbb;
#pragma unroll
        for (int t9 = 0; t9 < 9; t9++) {
            float m = (w9[t9] * gamma + beta) * wsc;
            Wm[(((size_t)(n * 9 + t9) * C_ + co) << 7) + ci] = f2bf(m);
        }
    }
}

// ---------------------------------------------------------------------------
// WMMA conv (9 taps, halo-padded input -> branchless) + bias + noise + lrelu
// Block = 128 thr (4 waves). Wave tile: 16 co x 32 pix (two 16x16 WMMA tiles,
// A fragment reused for both). Block covers 32 co x 64 pix of one output row.
// Grid: n(32) x co-pair(4) x h(64) = 8192 blocks.
// ---------------------------------------------------------------------------
union Frag { v16bf v; u32x4 q[2]; };

__global__ void __launch_bounds__(128) k_conv(const unsigned short* __restrict__ xTp,
                                              const unsigned short* __restrict__ Wm,
                                              const float* __restrict__ bi,
                                              const float* __restrict__ nzw,
                                              const float* __restrict__ noise,
                                              float* __restrict__ y) {
    int b = blockIdx.x;
    int h   = b & 63;
    int cop = (b >> 6) & 3;
    int n   = b >> 8;
    int tid  = threadIdx.x;
    int lane = tid & 31;
    int wave = tid >> 5;
    int half = lane >> 4;
    int l16  = lane & 15;
    int cot  = (cop << 1) + (wave >> 1);
    int wseg = wave & 1;
    int w0   = wseg << 5;                       // 32-pixel segment base
    int cob  = cot << 4;

    v8f acc0 = {0.f,0.f,0.f,0.f,0.f,0.f,0.f,0.f};
    v8f acc1 = {0.f,0.f,0.f,0.f,0.f,0.f,0.f,0.f};

#pragma unroll
    for (int kh = 0; kh < 3; kh++) {
        // padded source row: hp = (h + kh - 1) + 1 = h + kh  (always in [0,66))
        const unsigned short* xrow = xTp + (((size_t)(n * HP_ + h + kh) * WP_) << 7);
#pragma unroll
        for (int kw = 0; kw < 3; kw++) {
            // padded source col for lane pixel: wp = w0 + l16 + kw (in [0,66))
            const unsigned short* bb0 = xrow + (((size_t)(w0 + l16 + kw)) << 7);
            const unsigned short* bb1 = bb0 + (16 << 7);
            const unsigned short* arow =
                Wm + (((size_t)((n * 9 + kh * 3 + kw) * C_ + cob + l16)) << 7);
#pragma unroll
            for (int c4 = 0; c4 < 4; c4++) {
                int ci0 = c4 << 5;
                Frag a, b0f, b1f;
                // A lane layout: K = ci0 + half*16 + j (32 contiguous bytes)
                a.q[0] = *(const u32x4*)(arow + ci0 + (half << 4));
                a.q[1] = *(const u32x4*)(arow + ci0 + (half << 4) + 8);
                // B lane layout: K = ci0 + half*16 + j (32 contiguous bytes)
                b0f.q[0] = *(const u32x4*)(bb0 + ci0 + (half << 4));
                b0f.q[1] = *(const u32x4*)(bb0 + ci0 + (half << 4) + 8);
                b1f.q[0] = *(const u32x4*)(bb1 + ci0 + (half << 4));
                b1f.q[1] = *(const u32x4*)(bb1 + ci0 + (half << 4) + 8);
                acc0 = __builtin_amdgcn_wmma_f32_16x16x32_bf16(
                           false, a.v, false, b0f.v, (short)0, acc0, false, false);
                acc1 = __builtin_amdgcn_wmma_f32_16x16x32_bf16(
                           false, a.v, false, b1f.v, (short)0, acc1, false, false);
            }
        }
    }

    int p0 = (h << 6) + w0 + l16;
    int p1 = p0 + 16;
    float nz0 = noise[(size_t)n * HW_ + p0];
    float nz1 = noise[(size_t)n * HW_ + p1];
#pragma unroll
    for (int r = 0; r < 8; r++) {
        int co = cob + r + (half << 3);
        float bv = bi[(n << 7) + co];
        float nw = nzw[co] * 0.125f;            // sqrt(2/128) = 0.125
        float v0 = acc0[r] + bv + nw * nz0;
        float v1 = acc1[r] + bv + nw * nz1;
        v0 = (v0 > 0.f) ? v0 : 0.2f * v0;
        v1 = (v1 > 0.f) ? v1 : 0.2f * v1;
        size_t base = ((size_t)((n << 7) + co)) << 12;
        y[base + p0] = v0;
        y[base + p1] = v1;
    }
}

// ---------------------------------------------------------------------------
// Per-(n,c) mean / rstd over 4096 pixels
// ---------------------------------------------------------------------------
__global__ void k_stats(const float* __restrict__ y, float2* __restrict__ st) {
    __shared__ float ss[256], sq[256];
    int nc = blockIdx.x;
    const float* p = y + ((size_t)nc << 12);
    float s = 0.f, q = 0.f;
    for (int i = threadIdx.x; i < HW_; i += 256) { float v = p[i]; s += v; q += v * v; }
    ss[threadIdx.x] = s; sq[threadIdx.x] = q;
    __syncthreads();
    for (int o = 128; o > 0; o >>= 1) {
        if (threadIdx.x < o) { ss[threadIdx.x] += ss[threadIdx.x + o]; sq[threadIdx.x] += sq[threadIdx.x + o]; }
        __syncthreads();
    }
    if (threadIdx.x == 0) {
        float mean = ss[0] * (1.f / 4096.f);
        float var  = sq[0] * (1.f / 4096.f) - mean * mean;
        float2 r; r.x = mean; r.y = rsqrtf(var + 1e-5f);
        st[nc] = r;
    }
}

// ---------------------------------------------------------------------------
// AdaIN apply (layer 1): normalize y, write bf16 into padded NHWC interior
// ---------------------------------------------------------------------------
__global__ void k_apply1(const float* __restrict__ y, const float2* __restrict__ st,
                         const float* __restrict__ s1, unsigned short* __restrict__ xTp) {
    __shared__ float tile[C_ * W_];
    int b = blockIdx.x; int n = b >> 6, h = b & 63;
    for (int i = threadIdx.x; i < C_ * W_; i += 256) {
        int c = i >> 6, w = i & 63;
        float2 mr = st[(n << 7) + c];
        float g  = s1[(n << 8) + c];
        float be = s1[(n << 8) + 128 + c];
        float v = y[(((size_t)((n << 7) + c)) << 12) + (h << 6) + w];
        tile[(c << 6) + w] = g * (v - mr.x) * mr.y + be;
    }
    __syncthreads();
    unsigned short* dst = xTp + (((size_t)(n * HP_ + h + 1) * WP_ + 1) << 7);
    for (int i = threadIdx.x; i < C_ * W_; i += 256) {
        int w = i >> 7, c = i & 127;
        dst[((size_t)w << 7) + c] = f2bf(tile[(c << 6) + w]);
    }
}

// ---------------------------------------------------------------------------
// AdaIN apply (layer 2): normalize y2 -> d_out (N,C,H,W) f32, elementwise
// ---------------------------------------------------------------------------
__global__ void k_apply2(const float* __restrict__ y, const float2* __restrict__ st,
                         const float* __restrict__ s2, float* __restrict__ out) {
    size_t idx = (size_t)blockIdx.x * 256 + threadIdx.x;
    int nc = (int)(idx >> 12);
    int n = nc >> 7, c = nc & 127;
    float2 mr = st[nc];
    float g  = s2[(n << 8) + c];
    float be = s2[(n << 8) + 128 + c];
    out[idx] = g * (y[idx] - mr.x) * mr.y + be;
}

extern "C" void kernel_launch(void* const* d_in, const int* in_sizes, int n_in,
                              void* d_out, int out_size, void* d_ws, size_t ws_size,
                              hipStream_t stream) {
    const float* x     = (const float*)d_in[0];
    const float* style = (const float*)d_in[1];
    const float* noise = (const float*)d_in[2];
    const float* task  = (const float*)d_in[3];
    const float* W1    = (const float*)d_in[4];
    const float* b1    = (const float*)d_in[5];
    const float* t1_w  = (const float*)d_in[6];
    const float* t1_b  = (const float*)d_in[7];
    const float* tb1_w = (const float*)d_in[8];
    const float* tb1_b = (const float*)d_in[9];
    const float* nz1_w = (const float*)d_in[10];
    const float* ad1_w = (const float*)d_in[11];
    const float* ad1_b = (const float*)d_in[12];
    const float* W2    = (const float*)d_in[13];
    const float* b2    = (const float*)d_in[14];
    const float* t2_w  = (const float*)d_in[15];
    const float* t2_b  = (const float*)d_in[16];
    const float* tb2_w = (const float*)d_in[17];
    const float* tb2_b = (const float*)d_in[18];
    const float* nz2_w = (const float*)d_in[19];
    const float* ad2_w = (const float*)d_in[20];
    const float* ad2_b = (const float*)d_in[21];

    char* ws = (char*)d_ws;
    unsigned short* xTp = (unsigned short*)ws;                         // 35,684,352 B padded NHWC bf16
    unsigned short* Wm  = (unsigned short*)(ws + (size_t)37748736);    // 9,437,184 B (reused both layers)
    float*  y     = (float*)(ws + (size_t)50331648);                   // 67,108,864 B (reused)
    float*  bi1   = (float*)(ws + (size_t)117440512);
    float*  bi2   = bi1 + 4096;
    float*  s1    = bi2 + 4096;
    float*  s2    = s1 + 8192;
    float2* stats = (float2*)(ws + (size_t)117571584);                 // 32 KB

    const int xtp16 = XTP_ELEMS * 2 / 16;                              // uint4 count
    k_zero<<<(xtp16 + 255) / 256, 256, 0, stream>>>((u32x4*)xTp, xtp16);
    k_convert_x<<<N_ * H_, 256, 0, stream>>>(x, xTp);
    k_small<<<96, 256, 0, stream>>>(task, style, b1, tb1_w, tb1_b, b2, tb2_w, tb2_b,
                                    ad1_w, ad1_b, ad2_w, ad2_b, bi1, bi2, s1, s2);
    // ---- layer 1 ----
    k_modw<<<64, 256, 0, stream>>>(task, t1_w, t1_b, W1, Wm);
    k_conv<<<N_ * 4 * H_, 128, 0, stream>>>(xTp, Wm, bi1, nz1_w, noise, y);
    k_stats<<<N_ * C_, 256, 0, stream>>>(y, stats);
    k_apply1<<<N_ * H_, 256, 0, stream>>>(y, stats, s1, xTp);
    // ---- layer 2 ----
    k_modw<<<64, 256, 0, stream>>>(task, t2_w, t2_b, W2, Wm);
    k_conv<<<N_ * 4 * H_, 128, 0, stream>>>(xTp, Wm, bi2, nz2_w, noise, y);
    k_stats<<<N_ * C_, 256, 0, stream>>>(y, stats);
    k_apply2<<<N_ * C_ * HW_ / 256, 256, 0, stream>>>(y, stats, s2, (float*)d_out);
}
